// CharacterNet_38457137168845
// MI455X (gfx1250) — compile-verified, matrix operand order checked by reading, not run on hardware
//
#include <hip/hip_runtime.h>

#define LAYERS 13
#define BATCH  32
#define SEQ    512
#define HID    768
#define TOK    256
#define MROWS  (BATCH * TOK)   // 8192

// LDS layout for the staged B tile: 64 rows, each 768 halves (1536 B) padded
// to 1552 B so b128 fragment reads rotate across banks (1552 % 256 = 16).
#define BROWSTRIDE 1552
#define SMEM_BYTES (64 * BROWSTRIDE)   // 99,328 B < 320 KB WGP LDS

typedef __attribute__((ext_vector_type(16))) _Float16 v16h;
typedef __attribute__((ext_vector_type(8)))  _Float16 v8h;
typedef __attribute__((ext_vector_type(8)))  float    v8f;

union Frag16 { v16h v; v8h h[2]; };

// ---------------------------------------------------------------------------
// Kernel 1: segment mean pooling.
// One block per (b, t) segment; 256 threads each cover 3 of the 768 hidden
// channels. Writes fp32 means to the second output (original_token_embeddings)
// and an f16 copy to workspace as the GEMM A matrix.
// ---------------------------------------------------------------------------
__global__ void pool_kernel(const float* __restrict__ layers,
                            const int*   __restrict__ token_map,
                            const int*   __restrict__ layer_index,
                            float*       __restrict__ emb_out,   // (MROWS, HID) f32
                            _Float16*    __restrict__ emb_h)     // (MROWS, HID) f16
{
    const int bt = blockIdx.x;
    const int b  = bt / TOK;
    const int t  = bt - b * TOK;
    const int li = layer_index[0];
    const int start = token_map[b * (TOK + 1) + t];
    const int end   = token_map[b * (TOK + 1) + t + 1];
    const float* base = layers + ((size_t)li * BATCH + b) * (size_t)SEQ * HID;
    const float inv = 1.0f / (float)(end - start);

    const int h = threadIdx.x;
    float s0 = 0.f, s1 = 0.f, s2 = 0.f;
    for (int r = start; r < end; ++r) {
        const float* row = base + (size_t)r * HID;
        s0 += row[h];
        s1 += row[h + 256];
        s2 += row[h + 512];
    }
    const float m0 = s0 * inv, m1 = s1 * inv, m2 = s2 * inv;
    const size_t o = (size_t)bt * HID;
    emb_out[o + h]       = m0;
    emb_out[o + h + 256] = m1;
    emb_out[o + h + 512] = m2;
    emb_h[o + h]         = (_Float16)m0;
    emb_h[o + h + 256]   = (_Float16)m1;
    emb_h[o + h + 512]   = (_Float16)m2;
}

// ---------------------------------------------------------------------------
// Kernel 2: fc_w f32 -> f16 (768*768 elements).
// ---------------------------------------------------------------------------
__global__ void cvt_w_kernel(const float* __restrict__ w, _Float16* __restrict__ wh)
{
    const int i = blockIdx.x * blockDim.x + threadIdx.x;
    wh[i] = (_Float16)w[i];
}

// ---------------------------------------------------------------------------
// Kernel 3: rep[M,N] = A[M,K] * W[N,K]^T + bias via v_wmma_f32_16x16x32_f16.
//
// Block: 8 waves, each owning a 32(M) x 64(N) tile -> block = 256 M x 64 N.
// The block's 64x768 f16 B tile is staged in LDS once (96 KB, padded rows);
// each wave then runs the K loop with:
//   - two A fragments (rows mbase.. and mbase+16..) register double-buffered
//     from global (plus a global_prefetch_b8 one K-step further ahead),
//   - four B fragments per K-step from LDS (ds_load_b128), each feeding TWO
//     WMMAs (a0 x b, a1 x b) -> 8 WMMAs per K-step, halving ds-per-wmma.
//
// Fragment layouts (wave32):
//   A 16x32 f16: lane L<16 -> row base+L, halves k0+0..7 / k0+16..23;
//                lane L>=16 -> row base+L-16, halves k0+8..15 / k0+24..31.
//   B 32x16 f16 mirrors this with column n; W is (n,k) row-major so the
//   per-lane chunks are contiguous 16-byte runs in both global and LDS.
//   C/D: VGPR r, lanes 0..15 -> M=r; lanes 16..31 -> M=r+8; N = lane%16.
// ---------------------------------------------------------------------------
__global__ void gemm_kernel(const _Float16* __restrict__ A,    // M x K
                            const _Float16* __restrict__ Wh,   // N x K
                            const float*    __restrict__ bias, // N
                            float*          __restrict__ rep)  // M x N
{
    extern __shared__ char smem[];

    const int K = HID, N = HID;
    const int tid    = threadIdx.x;
    const int lane   = tid & 31;
    const int wave   = tid >> 5;
    const int mbase  = blockIdx.x * 256 + wave * 32;
    const int nbase  = blockIdx.y * 64;
    const int l15    = lane & 15;
    const int hiHalf = (lane >> 4) & 1;

    // ---- cooperative stage of the 64 x 1536B B tile into LDS ----
    {
        const char* gB = (const char*)(Wh + (size_t)nbase * K);
#pragma unroll
        for (int i = 0; i < 24; ++i) {
            const int chunk = tid + i * 256;        // 0..6143 16-byte chunks
            const int row   = chunk / 96;           // 96 chunks per 1536B row
            const int col   = chunk - row * 96;
            v8h v = *(const v8h*)(gB + (size_t)row * (K * 2) + col * 16);
            *(v8h*)(smem + row * BROWSTRIDE + col * 16) = v;
        }
    }
    __syncthreads();

    const _Float16* aRow0 = A + (size_t)(mbase + l15) * K + hiHalf * 8;
    const _Float16* aRow1 = aRow0 + (size_t)16 * K;
    const char*     bBase = smem + l15 * BROWSTRIDE + hiHalf * 16;

    v8f acc[2][4];
#pragma unroll
    for (int i = 0; i < 2; ++i)
#pragma unroll
        for (int s = 0; s < 4; ++s)
            acc[i][s] = (v8f){};

    Frag16 aCur0, aCur1;
    aCur0.h[0] = *(const v8h*)(aRow0);
    aCur0.h[1] = *(const v8h*)(aRow0 + 16);
    aCur1.h[0] = *(const v8h*)(aRow1);
    aCur1.h[1] = *(const v8h*)(aRow1 + 16);

    for (int k0 = 0; k0 < K; k0 += 32) {
        // prefetch next A fragments (uniform branch; EXEC stays all-ones)
        Frag16 aNext0 = aCur0, aNext1 = aCur1;
        if (k0 + 32 < K) {
            aNext0.h[0] = *(const v8h*)(aRow0 + k0 + 32);
            aNext0.h[1] = *(const v8h*)(aRow0 + k0 + 48);
            aNext1.h[0] = *(const v8h*)(aRow1 + k0 + 32);
            aNext1.h[1] = *(const v8h*)(aRow1 + k0 + 48);
            // nudge the K-step after that toward L2/L0 (global_prefetch_b8)
            __builtin_prefetch(aRow0 + k0 + 64, 0, 1);
            __builtin_prefetch(aRow1 + k0 + 64, 0, 1);
        }

        const char* bK = bBase + k0 * 2;
#pragma unroll
        for (int s = 0; s < 4; ++s) {
            Frag16 b;
            b.h[0] = *(const v8h*)(bK + s * 16 * BROWSTRIDE);
            b.h[1] = *(const v8h*)(bK + s * 16 * BROWSTRIDE + 32);
            acc[0][s] = __builtin_amdgcn_wmma_f32_16x16x32_f16(
                false, aCur0.v, false, b.v, (short)0, acc[0][s], false, false);
            acc[1][s] = __builtin_amdgcn_wmma_f32_16x16x32_f16(
                false, aCur1.v, false, b.v, (short)0, acc[1][s], false, false);
        }
        aCur0 = aNext0;
        aCur1 = aNext1;
    }

    // ---- epilogue: bias add + store ----
    const float bv[4] = { bias[nbase + l15],      bias[nbase + 16 + l15],
                          bias[nbase + 32 + l15], bias[nbase + 48 + l15] };
    const int mrow0 = mbase + hiHalf * 8;        // rows for acc[0][*]
    const int mrow1 = mbase + 16 + hiHalf * 8;   // rows for acc[1][*]

#pragma unroll
    for (int r = 0; r < 8; ++r) {
        float* o0 = rep + (size_t)(mrow0 + r) * N + nbase + l15;
        float* o1 = rep + (size_t)(mrow1 + r) * N + nbase + l15;
#pragma unroll
        for (int s = 0; s < 4; ++s) {
            o0[16 * s] = acc[0][s][r] + bv[s];
            o1[16 * s] = acc[1][s][r] + bv[s];
        }
    }
}

// ---------------------------------------------------------------------------
extern "C" void kernel_launch(void* const* d_in, const int* in_sizes, int n_in,
                              void* d_out, int out_size, void* d_ws, size_t ws_size,
                              hipStream_t stream)
{
    const float* layers      = (const float*)d_in[0];
    // d_in[1] = input_mask (all ones; pooling fully determined by token_map)
    const int*   token_map   = (const int*)d_in[2];
    const float* fc_w        = (const float*)d_in[3];
    const float* fc_b        = (const float*)d_in[4];
    const int*   layer_index = (const int*)d_in[5];

    float* rep = (float*)d_out;                       // output 0: (M, H)
    float* emb = rep + (size_t)MROWS * HID;           // output 1: (M, H)

    _Float16* embh = (_Float16*)d_ws;                 // M*H f16  (12.6 MB)
    _Float16* wh   = embh + (size_t)MROWS * HID;      // H*H f16  (1.2 MB)

    pool_kernel<<<MROWS, 256, 0, stream>>>(layers, token_map, layer_index, emb, embh);
    cvt_w_kernel<<<(HID * HID) / 256, 256, 0, stream>>>(fc_w, wh);

    dim3 grid(MROWS / 256, HID / 64);                 // 32 x 12 blocks, 8 waves each
    gemm_kernel<<<grid, 256, SMEM_BYTES, stream>>>(embh, wh, fc_b, rep);
}
